// GPTLanguageModel_72748156059821
// MI455X (gfx1250) — compile-verified, matrix operand order checked by reading, not run on hardware
//
#include <hip/hip_runtime.h>
#include <hip/hip_bf16.h>

// ---------------------------------------------------------------------------
// GPT forward for gfx1250 (MI455X). All matmuls via v_wmma_f32_16x16x32_f16.
// GEMM tiles staged global->LDS with GLOBAL_LOAD_ASYNC_TO_LDS_B128 (ASYNCcnt)
// when the toolchain exposes the builtin; otherwise plain loads + prefetch.
// fp32->f16 conversion folded into WMMA fragment construction (memory-bound
// problem: the 524MB fp32 logits stream dominates; weights read once).
// ---------------------------------------------------------------------------

typedef _Float16 half_t;
typedef __attribute__((ext_vector_type(16))) _Float16 v16h;
typedef __attribute__((ext_vector_type(8)))  float    v8f;
typedef int i128 __attribute__((__vector_size__(16)));   // 16-byte transfer unit

#if defined(__has_builtin)
#  if __has_builtin(__builtin_amdgcn_global_load_async_to_lds_b128)
#    define HAVE_ASYNC_LDS 1
#  endif
#endif
#ifndef HAVE_ASYNC_LDS
#  define HAVE_ASYNC_LDS 0
#endif

__device__ __forceinline__ void async_copy_b128(const float* g, float* l) {
#if HAVE_ASYNC_LDS
  // Signature (from clang diagnostic): param0 = AS1 (global) i128*,
  // param1 = LDS i128*, then i32 offset, i32 cpol.
  __builtin_amdgcn_global_load_async_to_lds_b128(
      (__attribute__((address_space(1))) i128*)(float*)g,
      (__attribute__((address_space(3))) i128*)l,
      0, 0);
#else
  *(float4*)l = *(const float4*)g;
#endif
}

__device__ __forceinline__ void wait_async() {
#if HAVE_ASYNC_LDS
#  if __has_builtin(__builtin_amdgcn_s_wait_asynccnt)
  __builtin_amdgcn_s_wait_asynccnt(0);
#  else
  asm volatile("s_wait_asynccnt 0" ::: "memory");
#  endif
#endif
}

// ---------------------------------------------------------------------------
// Embedding: x[b,t,:] = tok_emb[tokens[b,t]] + pet[t]
// ---------------------------------------------------------------------------
__global__ void embed_kernel(const int* __restrict__ tokens,
                             const float* __restrict__ tok_emb,
                             const float* __restrict__ pet,
                             float* __restrict__ x,
                             int BT, int E, int T) {
  int idx = blockIdx.x * blockDim.x + threadIdx.x;
  int total = BT * E;
  int stride = gridDim.x * blockDim.x;
  for (; idx < total; idx += stride) {
    int rt = idx / E;
    int e  = idx - rt * E;
    int t  = rt % T;
    x[idx] = tok_emb[(size_t)tokens[rt] * E + e] + pet[(size_t)t * E + e];
  }
}

// ---------------------------------------------------------------------------
// LayerNorm: one 256-thread block per row
// ---------------------------------------------------------------------------
__global__ __launch_bounds__(256)
void layernorm_kernel(const float* __restrict__ in,
                      const float* __restrict__ g,
                      const float* __restrict__ b,
                      float* __restrict__ out, int ncols) {
  __shared__ float red[256];
  int row = blockIdx.x, tid = threadIdx.x;
  const float* ip = in + (size_t)row * ncols;
  float s = 0.f;
  for (int c = tid; c < ncols; c += 256) s += ip[c];
  red[tid] = s; __syncthreads();
  for (int st = 128; st > 0; st >>= 1) {
    if (tid < st) red[tid] += red[tid + st];
    __syncthreads();
  }
  float mean = red[0] / (float)ncols; __syncthreads();
  float vs = 0.f;
  for (int c = tid; c < ncols; c += 256) { float d = ip[c] - mean; vs += d * d; }
  red[tid] = vs; __syncthreads();
  for (int st = 128; st > 0; st >>= 1) {
    if (tid < st) red[tid] += red[tid + st];
    __syncthreads();
  }
  float inv = rsqrtf(red[0] / (float)ncols + 1e-5f);
  float* op = out + (size_t)row * ncols;
  for (int c = tid; c < ncols; c += 256)
    op[c] = (ip[c] - mean) * inv * g[c] + b[c];
}

// ---------------------------------------------------------------------------
// WMMA GEMM: C[M,N] = A[M,K] @ B[K,N] (+bias, +residual, +relu)
// Workgroup tile 32x128, BK=32; 8 waves in a 2x4 grid, each wave computes a
// 16x32 strip (1 A-fragment reused across 2 WMMAs). Tiles staged to LDS in
// fp32 via async-DMA (B128 per lane), converted to f16 at fragment build.
// BLAYOUT 0: B row-major [K,N].  BLAYOUT 1: B is (H,E,HS)=(*,K,64) blocked
// (QKV weights); head index = global_col/64 (BN=128 spans 2 heads).
// ---------------------------------------------------------------------------
template<int BLAYOUT, bool BIAS, bool RESID, bool RELU>
__global__ __launch_bounds__(256)
void gemm_wmma_kernel(const float* __restrict__ A,
                      const float* __restrict__ Bw,
                      const float* __restrict__ bias,
                      const float* __restrict__ resid,
                      float* __restrict__ C, int M, int N, int K) {
  __shared__ float Asf[32][32];    // 4 KB
  __shared__ float Bsf[32][128];   // 16 KB
  const int bx = blockIdx.x, by = blockIdx.y;
  const int tid = threadIdx.x;
  const int wid = tid >> 5, lane = tid & 31;
  const int wr = wid >> 2, wc = wid & 3;          // 2x4 wave grid
  const int halfI = lane >> 4, ml = lane & 15;

  // Per-thread staging coordinates (16B = 4 floats per async transfer)
  const int am  = tid >> 3;            // A row 0..31
  const int ak  = (tid & 7) * 4;       // A col 0..28
  v8f acc0 = {}, acc1 = {};

  for (int k0 = 0; k0 < K; k0 += 32) {
    // ---- stage A (32x32 fp32): one B128 per thread --------------------
    async_copy_b128(&A[(size_t)(by * 32 + am) * K + (k0 + ak)], &Asf[am][ak]);
    // ---- stage B (32x128 fp32): four B128 per thread ------------------
#pragma unroll
    for (int i = 0; i < 4; ++i) {
      int j  = tid + i * 256;          // 0..1023
      int bk = j >> 5;                 // 0..31
      int bn = (j & 31) * 4;           // 0..124
      const float* src;
      if (BLAYOUT == 0) {
        src = &Bw[(size_t)(k0 + bk) * N + (bx * 128 + bn)];
      } else {
        int col = bx * 128 + bn;       // 4 consecutive cols stay in one head
        int hh = col >> 6, f = col & 63;
        src = &Bw[((size_t)hh * K + (k0 + bk)) * 64 + f];
      }
      async_copy_b128(src, &Bsf[bk][bn]);
    }
#if !HAVE_ASYNC_LDS
    if (k0 + 32 < K)                   // prefetch next A chunk (global_prefetch_b8)
      __builtin_prefetch(&A[(size_t)(by * 32 + am) * K + (k0 + 32 + ak)], 0, 1);
#endif
    wait_async();
    __syncthreads();

    // ---- fragments (fp32 LDS -> f16 regs, ISA 16-bit A/B layouts) -----
    v16h af, bf0, bf1;
#pragma unroll
    for (int e = 0; e < 16; ++e) {
      int kk = ((e < 8) ? 0 : 16) + halfI * 8 + (e & 7);
      af[e]  = (half_t)Asf[wr * 16 + ml][kk];
      bf0[e] = (half_t)Bsf[kk][wc * 32 + ml];
      bf1[e] = (half_t)Bsf[kk][wc * 32 + 16 + ml];
    }
    acc0 = __builtin_amdgcn_wmma_f32_16x16x32_f16(false, af, false, bf0,
                                                  (short)0, acc0, false, false);
    acc1 = __builtin_amdgcn_wmma_f32_16x16x32_f16(false, af, false, bf1,
                                                  (short)0, acc1, false, false);
    __syncthreads();
  }

  const int row0 = by * 32 + wr * 16, col0 = bx * 128 + wc * 32;
#pragma unroll
  for (int r = 0; r < 8; ++r) {
    int gm = row0 + r + 8 * halfI;     // C layout: lanes 16-31 hold M=r+8
#pragma unroll
    for (int t = 0; t < 2; ++t) {
      int gn = col0 + t * 16 + ml;
      float val = (t == 0) ? acc0[r] : acc1[r];
      if (BIAS)  val += bias[gn];
      if (RESID) val += resid[(size_t)gm * N + gn];
      if (RELU)  val = fmaxf(val, 0.f);
      C[(size_t)gm * N + gn] = val;
    }
  }
}

// ---------------------------------------------------------------------------
// Flash-style causal attention. One 32-thread wave per (b, h, 16-query block).
// Key blocks of 32: S(16x32)=Q Kt via 4 WMMAs, online softmax, O += P V via
// 4 WMMAs. q/k/v/att stored [B,T,H*HS] (head-concatenated columns).
// ---------------------------------------------------------------------------
__global__ __launch_bounds__(32)
void attn_kernel(const float* __restrict__ Q, const float* __restrict__ Kb,
                 const float* __restrict__ Vb, float* __restrict__ Ob,
                 int B, int T, int H, int HS) {
  __shared__ half_t Qs[16][64];
  __shared__ half_t Ks[32][64];
  __shared__ half_t Vs[32][64];
  __shared__ half_t Ps[16][32];
  const int nQB = T / 16;
  const int qb = blockIdx.x % nQB;
  const int hh = (blockIdx.x / nQB) % H;
  const int bb = blockIdx.x / (nQB * H);
  const int lane = threadIdx.x;
  const int halfI = lane >> 4, ml = lane & 15;
  const int E = H * HS;
  const float scale = rsqrtf((float)HS);

  for (int i = lane; i < 16 * 64; i += 32) {
    int r = i >> 6, c = i & 63;
    Qs[r][c] = (half_t)(Q[((size_t)bb * T + qb * 16 + r) * E + hh * 64 + c] * scale);
  }
  float m_i[8], l_i[8];
  v8f o0 = {}, o1 = {}, o2 = {}, o3 = {};
#pragma unroll
  for (int r = 0; r < 8; ++r) { m_i[r] = -__builtin_inff(); l_i[r] = 0.f; }

  const int jmax = (qb * 16 + 15) / 32;
  for (int j = 0; j <= jmax; ++j) {
    for (int i = lane; i < 32 * 64; i += 32) {
      int r = i >> 6, c = i & 63;
      size_t base = ((size_t)bb * T + j * 32 + r) * E + hh * 64 + c;
      Ks[r][c] = (half_t)Kb[base];
      Vs[r][c] = (half_t)Vb[base];
    }
    __syncthreads();
    // S = Q Kt  (two 16-col key tiles, K-dim = 64 in two chunks)
    v8f s0 = {}, s1 = {};
#pragma unroll
    for (int dc = 0; dc < 64; dc += 32) {
      v16h aq, bk0, bk1;
#pragma unroll
      for (int e = 0; e < 16; ++e) {
        int kk = ((e < 8) ? 0 : 16) + halfI * 8 + (e & 7);
        aq[e]  = Qs[ml][dc + kk];
        bk0[e] = Ks[ml][dc + kk];        // Bt: column n=key, B[k][n]=K[key][d]
        bk1[e] = Ks[16 + ml][dc + kk];
      }
      s0 = __builtin_amdgcn_wmma_f32_16x16x32_f16(false, aq, false, bk0,
                                                  (short)0, s0, false, false);
      s1 = __builtin_amdgcn_wmma_f32_16x16x32_f16(false, aq, false, bk1,
                                                  (short)0, s1, false, false);
    }
    // Causal mask + online softmax (row stats reduced across 16-lane halves)
#pragma unroll
    for (int r = 0; r < 8; ++r) {
      int qg = qb * 16 + r + 8 * halfI;
      if (j * 32 + ml      > qg) s0[r] = -__builtin_inff();
      if (j * 32 + 16 + ml > qg) s1[r] = -__builtin_inff();
      float rm = fmaxf(s0[r], s1[r]);
#pragma unroll
      for (int off = 1; off < 16; off <<= 1)
        rm = fmaxf(rm, __shfl_xor(rm, off, 32));
      float mnew = fmaxf(m_i[r], rm);
      float corr = expf(m_i[r] - mnew);
      float p0 = expf(s0[r] - mnew);
      float p1 = expf(s1[r] - mnew);
      float rs = p0 + p1;
#pragma unroll
      for (int off = 1; off < 16; off <<= 1)
        rs += __shfl_xor(rs, off, 32);
      l_i[r] = l_i[r] * corr + rs;
      m_i[r] = mnew;
      o0[r] *= corr; o1[r] *= corr; o2[r] *= corr; o3[r] *= corr;
      Ps[r + 8 * halfI][ml]      = (half_t)p0;
      Ps[r + 8 * halfI][16 + ml] = (half_t)p1;
    }
    __syncthreads();
    // O += P(16x32) @ V(32x64)
    {
      v16h ap, bv0, bv1, bv2, bv3;
#pragma unroll
      for (int e = 0; e < 16; ++e) {
        int kk = ((e < 8) ? 0 : 16) + halfI * 8 + (e & 7);
        ap[e]  = Ps[ml][kk];
        bv0[e] = Vs[kk][ 0 + ml];
        bv1[e] = Vs[kk][16 + ml];
        bv2[e] = Vs[kk][32 + ml];
        bv3[e] = Vs[kk][48 + ml];
      }
      o0 = __builtin_amdgcn_wmma_f32_16x16x32_f16(false, ap, false, bv0, (short)0, o0, false, false);
      o1 = __builtin_amdgcn_wmma_f32_16x16x32_f16(false, ap, false, bv1, (short)0, o1, false, false);
      o2 = __builtin_amdgcn_wmma_f32_16x16x32_f16(false, ap, false, bv2, (short)0, o2, false, false);
      o3 = __builtin_amdgcn_wmma_f32_16x16x32_f16(false, ap, false, bv3, (short)0, o3, false, false);
    }
    __syncthreads();
  }
#pragma unroll
  for (int r = 0; r < 8; ++r) {
    float inv = 1.f / l_i[r];
    int t = qb * 16 + r + 8 * halfI;
    size_t base = ((size_t)bb * T + t) * E + hh * 64;
    Ob[base +  0 + ml] = o0[r] * inv;
    Ob[base + 16 + ml] = o1[r] * inv;
    Ob[base + 32 + ml] = o2[r] * inv;
    Ob[base + 48 + ml] = o3[r] * inv;
  }
}

// ---------------------------------------------------------------------------
// Loss: per-row log-softmax NLL, atomic mean into *loss
// ---------------------------------------------------------------------------
__global__ void zero_kernel(float* p) {
  if (threadIdx.x == 0 && blockIdx.x == 0) p[0] = 0.f;
}

__global__ __launch_bounds__(256)
void loss_kernel(const float* __restrict__ logits,
                 const int* __restrict__ targets,
                 float* __restrict__ loss, int nrows, int ncols) {
  __shared__ float red[256];
  int row = blockIdx.x, tid = threadIdx.x;
  const float* lp = logits + (size_t)row * ncols;
  float mx = -__builtin_inff();
  for (int c = tid; c < ncols; c += 256) mx = fmaxf(mx, lp[c]);
  red[tid] = mx; __syncthreads();
  for (int st = 128; st > 0; st >>= 1) {
    if (tid < st) red[tid] = fmaxf(red[tid], red[tid + st]);
    __syncthreads();
  }
  mx = red[0]; __syncthreads();
  float s = 0.f;
  for (int c = tid; c < ncols; c += 256) s += expf(lp[c] - mx);
  red[tid] = s; __syncthreads();
  for (int st = 128; st > 0; st >>= 1) {
    if (tid < st) red[tid] += red[tid + st];
    __syncthreads();
  }
  if (tid == 0) {
    float lse = mx + logf(red[0]);
    float tl = lp[targets[row]];
    atomicAdd(loss, (lse - tl) / (float)nrows);
  }
}

// ---------------------------------------------------------------------------
// Host driver
// ---------------------------------------------------------------------------
extern "C" void kernel_launch(void* const* d_in, const int* in_sizes, int n_in,
                              void* d_out, int out_size, void* d_ws, size_t ws_size,
                              hipStream_t stream) {
  (void)in_sizes; (void)n_in; (void)out_size; (void)ws_size;
  constexpr int B = 4, T = 1024, E = 768, H = 12, L = 6, V = 32000;
  constexpr int HS = 64, FF = 3072, M = B * T;

  const int*   tokens  = (const int*)  d_in[0];
  const int*   targets = (const int*)  d_in[1];
  const float* tok_emb = (const float*)d_in[2];
  const float* pet     = (const float*)d_in[3];
  const float* wq      = (const float*)d_in[4];
  const float* wk      = (const float*)d_in[5];
  const float* wv      = (const float*)d_in[6];
  const float* w_proj  = (const float*)d_in[7];
  const float* b_proj  = (const float*)d_in[8];
  const float* ln1_g   = (const float*)d_in[9];
  const float* ln1_b   = (const float*)d_in[10];
  const float* ln2_g   = (const float*)d_in[11];
  const float* ln2_b   = (const float*)d_in[12];
  const float* w_ff1   = (const float*)d_in[13];
  const float* b_ff1   = (const float*)d_in[14];
  const float* w_ff2   = (const float*)d_in[15];
  const float* b_ff2   = (const float*)d_in[16];
  const float* lnf_g   = (const float*)d_in[17];
  const float* lnf_b   = (const float*)d_in[18];
  const float* w_out   = (const float*)d_in[19];
  const float* b_out   = (const float*)d_in[20];

  // Workspace layout (fp32): ~126 MB total
  float* x  = (float*)d_ws;
  float* hb = x  + (size_t)M * E;   // LN output (reused per layer + final)
  float* qb = hb + (size_t)M * E;
  float* kb = qb + (size_t)M * E;
  float* vb = kb + (size_t)M * E;
  float* ab = vb + (size_t)M * E;   // attention output
  float* fb = ab + (size_t)M * E;   // FF1 output [M, FF]

  embed_kernel<<<2048, 256, 0, stream>>>(tokens, tok_emb, pet, x, M, E, T);

  const dim3 gE(E / 128, M / 32);
  const dim3 gF(FF / 128, M / 32);
  for (int l = 0; l < L; ++l) {
    const float* wq_l = wq + (size_t)l * H * E * HS;
    const float* wk_l = wk + (size_t)l * H * E * HS;
    const float* wv_l = wv + (size_t)l * H * E * HS;

    layernorm_kernel<<<M, 256, 0, stream>>>(x, ln1_g + (size_t)l * E,
                                            ln1_b + (size_t)l * E, hb, E);
    gemm_wmma_kernel<1, false, false, false><<<gE, 256, 0, stream>>>(
        hb, wq_l, nullptr, nullptr, qb, M, E, E);
    gemm_wmma_kernel<1, false, false, false><<<gE, 256, 0, stream>>>(
        hb, wk_l, nullptr, nullptr, kb, M, E, E);
    gemm_wmma_kernel<1, false, false, false><<<gE, 256, 0, stream>>>(
        hb, wv_l, nullptr, nullptr, vb, M, E, E);

    attn_kernel<<<B * H * (T / 16), 32, 0, stream>>>(qb, kb, vb, ab, B, T, H, HS);

    gemm_wmma_kernel<0, true, true, false><<<gE, 256, 0, stream>>>(
        ab, w_proj + (size_t)l * E * E, b_proj + (size_t)l * E, x, x, M, E, E);

    layernorm_kernel<<<M, 256, 0, stream>>>(x, ln2_g + (size_t)l * E,
                                            ln2_b + (size_t)l * E, hb, E);
    gemm_wmma_kernel<0, true, false, true><<<gF, 256, 0, stream>>>(
        hb, w_ff1 + (size_t)l * E * FF, b_ff1 + (size_t)l * FF, nullptr, fb,
        M, FF, E);
    gemm_wmma_kernel<0, true, true, false><<<gE, 256, 0, stream>>>(
        fb, w_ff2 + (size_t)l * FF * E, b_ff2 + (size_t)l * E, x, x, M, E, FF);
  }

  layernorm_kernel<<<M, 256, 0, stream>>>(x, lnf_g, lnf_b, hb, E);

  float* logits = (float*)d_out;
  float* loss   = logits + (size_t)M * V;
  const dim3 gV(V / 128, M / 32);
  gemm_wmma_kernel<0, true, false, false><<<gV, 256, 0, stream>>>(
      hb, w_out, b_out, nullptr, logits, M, V, E);

  zero_kernel<<<1, 1, 0, stream>>>(loss);
  loss_kernel<<<M, 256, 0, stream>>>(logits, targets, loss, M, V);
}
